// PVRCNNPlusPlusBackbone3D_57466662420864
// MI455X (gfx1250) — compile-verified, matrix operand order checked by reading, not run on hardware
//
#include <hip/hip_runtime.h>
#include <hip/hip_bf16.h>

typedef __attribute__((ext_vector_type(16))) _Float16 v16h;
typedef __attribute__((ext_vector_type(8)))  _Float16 v8h;
typedef __attribute__((ext_vector_type(8)))  float    v8f;

#define GRES 256

// ---------------------------------------------------------------- utilities

__global__ void fill_i32(int* __restrict__ p, int v, size_t n) {
    size_t i = (size_t)blockIdx.x * blockDim.x + threadIdx.x;
    size_t stride = (size_t)gridDim.x * blockDim.x;
    for (; i < n; i += stride) p[i] = v;
}

__global__ void zero_f32(float* __restrict__ p, int n) {
    int i = blockIdx.x * blockDim.x + threadIdx.x;
    if (i < n) p[i] = 0.f;
}

__global__ void scatter_grid(const int* __restrict__ coords, int* __restrict__ grid, int N) {
    int n = blockIdx.x * blockDim.x + threadIdx.x;
    if (n >= N) return;
    int x = coords[3 * n + 0];
    int y = coords[3 * n + 1];
    int z = coords[3 * n + 2];
    grid[((size_t)x * GRES + y) * GRES + z] = n;
}

// rulebook: nidx[t*Npad + n] = neighbor voxel index for tap t, or -1
__global__ void build_rulebook(const int* __restrict__ coords, const int* __restrict__ grid,
                               int* __restrict__ nidx, int N, int Npad) {
    int n = blockIdx.x * blockDim.x + threadIdx.x;
    if (n >= Npad) return;
    if (n >= N) {
        for (int t = 0; t < 27; ++t) nidx[(size_t)t * Npad + n] = -1;
        return;
    }
    int x = coords[3 * n + 0];
    int y = coords[3 * n + 1];
    int z = coords[3 * n + 2];
    for (int t = 0; t < 27; ++t) {        // t = ((di+1)*3 + (dj+1))*3 + (dk+1), matches _OFFS
        int xx = x + t / 9 - 1;
        int yy = y + (t / 3) % 3 - 1;
        int zz = z + t % 3 - 1;
        int v = -1;
        if ((unsigned)xx < (unsigned)GRES && (unsigned)yy < (unsigned)GRES &&
            (unsigned)zz < (unsigned)GRES)
            v = grid[((size_t)xx * GRES + yy) * GRES + zz];
        nidx[(size_t)t * Npad + n] = v;
    }
}

// convert input fp32 features -> padded f16 rows (stride ciPad, zero-padded)
__global__ void prep_x0(const float* __restrict__ F, _Float16* __restrict__ X,
                        int N, int ci, int ciPad, size_t total) {
    size_t tid = (size_t)blockIdx.x * blockDim.x + threadIdx.x;
    if (tid >= total) return;
    int p = (int)(tid % ciPad);
    size_t row = tid / ciPad;
    float v = (row < (size_t)N && p < ci) ? F[row * ci + p] : 0.f;
    X[tid] = (_Float16)v;
}

// pack W[27, ci, co] (fp32) into per-lane WMMA B fragments (f16):
// Wp[(((k*chunks + c)*tilesN + t)*32 + lane)*16 + j] = W[k][c*32 + K(lane,j)][t*16 + lane%16]
// with K(lane,j) = (j&7) + 8*(lane>>4) + (j>=8 ? 16 : 0)  (16-bit B 32x16 layout)
__global__ void prep_w(const float* __restrict__ W, _Float16* __restrict__ Wp,
                       int ci, int co, int chunks, int tilesN, int total) {
    int tid = blockIdx.x * blockDim.x + threadIdx.x;
    if (tid >= total) return;
    int j    = tid & 15;
    int lane = (tid >> 4) & 31;
    int rest = tid >> 9;
    int t = rest % tilesN;
    int c = (rest / tilesN) % chunks;
    int k = rest / (tilesN * chunks);
    int K = (j & 7) + 8 * (lane >> 4) + ((j >= 8) ? 16 : 0);
    int kk = c * 32 + K;
    float v = (kk < ci) ? W[((size_t)k * ci + kk) * co + t * 16 + (lane & 15)] : 0.f;
    Wp[tid] = (_Float16)v;
}

// ------------------------------------------------- gather-GEMM via WMMA f16

// One wave per 32-voxel M-block (2 x 16-row WMMA tiles, so every B fragment feeds
// 2 WMMAs) and TN co-tiles (each A fragment feeds TN WMMAs). gridDim.y splits the
// co dimension across waves for occupancy on the wide layers.
template <int CHUNKS, int TN>
__global__ __launch_bounds__(32) void sconv_wmma(const _Float16* __restrict__ X,
                                                 const _Float16* __restrict__ Wp,
                                                 const int* __restrict__ nidx,
                                                 float* __restrict__ Y,
                                                 int Npad, int co, int tilesNTotal) {
    constexpr int CIPAD = CHUNKS * 32;
    const int lane   = threadIdx.x;
    const int g      = lane >> 4;    // half-wave group
    const int r16    = lane & 15;
    const int mt     = blockIdx.x;
    const int ntBase = blockIdx.y * TN;
    const int row0   = mt * 32 + r16;
    const int row1   = row0 + 16;

    const v8f  vzero8 = {};
    const v16h hzero  = {};

    v8f acc[2][TN];
#pragma unroll
    for (int mb = 0; mb < 2; ++mb)
#pragma unroll
        for (int t = 0; t < TN; ++t) acc[mb][t] = vzero8;

    for (int k = 0; k < 27; ++k) {
        int idx0 = nidx[(size_t)k * Npad + row0];
        int idx1 = nidx[(size_t)k * Npad + row1];
        const _Float16* xr0 = X + (size_t)(idx0 < 0 ? 0 : idx0) * CIPAD + 8 * g;
        const _Float16* xr1 = X + (size_t)(idx1 < 0 ? 0 : idx1) * CIPAD + 8 * g;
        if (k + 1 < 27) {  // prefetch next tap's weight block (global_prefetch_b8)
            __builtin_prefetch(Wp + ((((size_t)(k + 1) * CHUNKS) * tilesNTotal + ntBase) * 32 +
                                     lane) * 16, 0, 3);
        }
#pragma unroll
        for (int c = 0; c < CHUNKS; ++c) {
            // A fragment: 16-bit A 16x32 layout. lane<16: K {0..7,16..23}; lane>=16: {8..15,24..31}
            v16h a0 = hzero;
            if (idx0 >= 0) {
                const _Float16* p = xr0 + c * 32;
                v8h lo = *(const v8h*)p;
                v8h hi = *(const v8h*)(p + 16);
                a0 = __builtin_shufflevector(lo, hi, 0, 1, 2, 3, 4, 5, 6, 7,
                                             8, 9, 10, 11, 12, 13, 14, 15);
            }
            v16h a1 = hzero;
            if (idx1 >= 0) {
                const _Float16* p = xr1 + c * 32;
                v8h lo = *(const v8h*)p;
                v8h hi = *(const v8h*)(p + 16);
                a1 = __builtin_shufflevector(lo, hi, 0, 1, 2, 3, 4, 5, 6, 7,
                                             8, 9, 10, 11, 12, 13, 14, 15);
            }
            const _Float16* wb =
                Wp + ((((size_t)k * CHUNKS + c) * tilesNTotal + ntBase) * 32 + lane) * 16;
#pragma unroll
            for (int t = 0; t < TN; ++t) {
                v16h bf = *(const v16h*)(wb + (size_t)t * 512);
                acc[0][t] = __builtin_amdgcn_wmma_f32_16x16x32_f16(
                    false, a0, false, bf, (short)0, acc[0][t], false, false);
                acc[1][t] = __builtin_amdgcn_wmma_f32_16x16x32_f16(
                    false, a1, false, bf, (short)0, acc[1][t], false, false);
            }
        }
    }
    // D layout: lanes 0-15 -> rows M=e (N=lane), lanes 16-31 -> rows M=8+e (N=lane-16)
#pragma unroll
    for (int t = 0; t < TN; ++t) {
        int col = (ntBase + t) * 16 + r16;
#pragma unroll
        for (int mb = 0; mb < 2; ++mb) {
#pragma unroll
            for (int e = 0; e < 8; ++e) {
                int orow = mt * 32 + mb * 16 + 8 * g + e;
                Y[(size_t)orow * co + col] = acc[mb][t][e];
            }
        }
    }
}

// ------------------------------------------------------------ batchnorm+relu

// coalesced partial sums: threads cover channels, 256/co row-groups per block,
// LDS reduce across row-groups, atomic add into per-channel accumulators
__global__ void bn_partial(const float* __restrict__ Y, float* __restrict__ sums,
                           int N, int co) {
    __shared__ float sh[256];
    __shared__ float sh2[256];
    const int R   = 256 / co;            // co in {16,32,64,128} divides 256
    const int ch  = threadIdx.x % co;
    const int sub = threadIdx.x / co;
    float s = 0.f, ss = 0.f;
    for (int r = blockIdx.x * R + sub; r < N; r += gridDim.x * R) {
        float v = Y[(size_t)r * co + ch];
        s += v;
        ss += v * v;
    }
    sh[threadIdx.x]  = s;
    sh2[threadIdx.x] = ss;
    __syncthreads();
    if (sub == 0) {
        for (int j = 1; j < R; ++j) {
            s  += sh[j * co + ch];
            ss += sh2[j * co + ch];
        }
        atomicAdd(&sums[ch], s);
        atomicAdd(&sums[co + ch], ss);
    }
}

__global__ void bn_finalize(const float* __restrict__ sums, const float* __restrict__ gg,
                            const float* __restrict__ bb, float* __restrict__ stats,
                            int N, int co) {
    int ch = blockIdx.x * blockDim.x + threadIdx.x;
    if (ch >= co) return;
    float m   = sums[ch] / (float)N;
    float var = sums[co + ch] / (float)N - m * m;
    float inv = rsqrtf(var + 1e-3f);
    float a   = gg[ch] * inv;
    stats[ch]      = a;                 // scale
    stats[co + ch] = bb[ch] - m * a;    // shift
}

__global__ void bn_relu_f16(const float* __restrict__ Y, const float* __restrict__ stats,
                            _Float16* __restrict__ Xn, int N, int co, int ciPadN,
                            size_t total) {
    size_t tid = (size_t)blockIdx.x * blockDim.x + threadIdx.x;
    if (tid >= total) return;
    int p = (int)(tid % ciPadN);
    size_t row = tid / ciPadN;
    float v = 0.f;
    if (row < (size_t)N && p < co) {
        v = fmaxf(Y[row * co + p] * stats[p] + stats[co + p], 0.f);
    }
    Xn[tid] = (_Float16)v;
}

__global__ void bn_relu_out(const float* __restrict__ Y, const float* __restrict__ stats,
                            float* __restrict__ out, int N, int co) {
    size_t tid = (size_t)blockIdx.x * blockDim.x + threadIdx.x;
    if (tid >= (size_t)N * co) return;
    int p = (int)(tid % co);
    out[tid] = fmaxf(Y[tid] * stats[p] + stats[co + p], 0.f);
}

// --------------------------------------------------------------------- host

extern "C" void kernel_launch(void* const* d_in, const int* in_sizes, int n_in,
                              void* d_out, int out_size, void* d_ws, size_t ws_size,
                              hipStream_t stream) {
    const float* feat   = (const float*)d_in[0];
    const int*   coords = (const int*)d_in[1];

    const int N    = in_sizes[1] / 3;
    const int Npad = (N + 63) & ~63;

    static const int ciArr[13] = {4, 16, 16, 32, 32, 32, 64, 64, 64, 64, 64, 64, 64};
    static const int coArr[13] = {16, 16, 32, 32, 32, 64, 64, 64, 64, 64, 64, 64, 128};
    // co-tiles per wave (remaining tiles split across gridDim.y)
    static const int tnArr[13] = {1, 1, 2, 2, 2, 2, 2, 2, 2, 2, 2, 2, 4};

    // workspace partition (all offsets 256B aligned)
    char* ws = (char*)d_ws;
    size_t off = 0;
    auto take = [&](size_t bytes) {
        void* p = ws + off;
        off += (bytes + 255) & ~(size_t)255;
        return p;
    };
    int*       grid  = (int*)take((size_t)GRES * GRES * GRES * 4);   // 67 MB
    int*       nidx  = (int*)take((size_t)27 * Npad * 4);
    _Float16*  Xa    = (_Float16*)take((size_t)Npad * 64 * 2);
    _Float16*  Xb    = (_Float16*)take((size_t)Npad * 64 * 2);
    float*     Y     = (float*)take((size_t)Npad * 128 * 4);
    _Float16*  Wp    = (_Float16*)take((size_t)27 * 64 * 128 * 2);
    float*     sums  = (float*)take(1024);
    float*     stats = (float*)take(1024);
    (void)ws_size;

    // 1) dense grid hash
    size_t cells = (size_t)GRES * GRES * GRES;
    fill_i32<<<8192, 256, 0, stream>>>(grid, -1, cells);
    scatter_grid<<<(N + 255) / 256, 256, 0, stream>>>(coords, grid, N);
    build_rulebook<<<(Npad + 255) / 256, 256, 0, stream>>>(coords, grid, nidx, N, Npad);

    // 2) input features -> padded f16
    {
        size_t total = (size_t)Npad * 32;
        prep_x0<<<(unsigned)((total + 255) / 256), 256, 0, stream>>>(feat, Xa, N, 4, 32, total);
    }

    _Float16* Xcur = Xa;
    _Float16* Xnxt = Xb;
    const int tilesM32 = Npad / 32;

    for (int l = 0; l < 13; ++l) {
        const int ci = ciArr[l], co = coArr[l];
        const int ciPad  = (ci <= 32) ? 32 : 64;
        const int chunks = ciPad / 32;
        const int tilesN = co / 16;
        const int tn     = tnArr[l];
        const int splits = tilesN / tn;

        const float* W = (const float*)d_in[2 + 3 * l];
        const float* g = (const float*)d_in[3 + 3 * l];
        const float* b = (const float*)d_in[4 + 3 * l];

        // pack weights into B fragments
        int wtotal = 27 * chunks * tilesN * 512;
        prep_w<<<(wtotal + 255) / 256, 256, 0, stream>>>(W, Wp, ci, co, chunks, tilesN, wtotal);

        // gather-GEMM
        dim3 cgrid(tilesM32, splits);
        if (chunks == 1 && tn == 1)
            sconv_wmma<1, 1><<<cgrid, 32, 0, stream>>>(Xcur, Wp, nidx, Y, Npad, co, tilesN);
        else if (chunks == 1 && tn == 2)
            sconv_wmma<1, 2><<<cgrid, 32, 0, stream>>>(Xcur, Wp, nidx, Y, Npad, co, tilesN);
        else if (chunks == 2 && tn == 2)
            sconv_wmma<2, 2><<<cgrid, 32, 0, stream>>>(Xcur, Wp, nidx, Y, Npad, co, tilesN);
        else
            sconv_wmma<2, 4><<<cgrid, 32, 0, stream>>>(Xcur, Wp, nidx, Y, Npad, co, tilesN);

        // batchnorm stats (coalesced partials + atomics + finalize)
        zero_f32<<<1, 256, 0, stream>>>(sums, 2 * co);
        bn_partial<<<128, 256, 0, stream>>>(Y, sums, N, co);
        bn_finalize<<<1, 128, 0, stream>>>(sums, g, b, stats, N, co);

        if (l < 12) {
            const int ciPadN = (co <= 32) ? 32 : 64;
            size_t total = (size_t)Npad * ciPadN;
            bn_relu_f16<<<(unsigned)((total + 255) / 256), 256, 0, stream>>>(
                Y, stats, Xnxt, N, co, ciPadN, total);
            _Float16* t = Xcur; Xcur = Xnxt; Xnxt = t;
        } else {
            size_t total = (size_t)N * co;
            bn_relu_out<<<(unsigned)((total + 255) / 256), 256, 0, stream>>>(
                Y, stats, (float*)d_out, N, co);
        }
    }
}
